// SAASR_73186242724544
// MI455X (gfx1250) — compile-verified, matrix operand order checked by reading, not run on hardware
//
#include <hip/hip_runtime.h>
#include <hip/hip_bf16.h>

typedef _Float16 half_t;
typedef __attribute__((ext_vector_type(16))) _Float16 v16h;
typedef __attribute__((ext_vector_type(8)))  _Float16 v8h;
typedef __attribute__((ext_vector_type(8)))  float    v8f;

#define DD    384
#define NHH   4
#define HDIM  96
#define FFD   1024
#define LENC  750
#define LPAD  768        /* LENC padded to multiple of 32 for WMMA K */
#define TAUD  1500
#define BB    8
#define TT    256
#define VOC   60
#define SSPK  8
#define NLAY  2
#define MROW  (BB * LENC)   /* 6000 */

// ---------------------------------------------------------------------------
// f32 -> f16 conversion
// ---------------------------------------------------------------------------
__global__ __launch_bounds__(256) void cvt_f16_kernel(const float* __restrict__ x,
                                                      half_t* __restrict__ y, long n)
{
    long i = (long)blockIdx.x * 256 + threadIdx.x;
    if (i < n) y[i] = (half_t)x[i];
}

// ---------------------------------------------------------------------------
// WMMA fragment load: per-lane halves live at K = k0+8g+[0..7] and
// k0+16+8g+[0..7]  ->  two contiguous 16-byte global_load_b128.
// ---------------------------------------------------------------------------
__device__ inline v16h load_frag(const half_t* __restrict__ p)
{
    v8h lo = *(const v8h*)(p);
    v8h hi = *(const v8h*)(p + 16);
    return __builtin_shufflevector(lo, hi, 0, 1, 2, 3, 4, 5, 6, 7,
                                   8, 9, 10, 11, 12, 13, 14, 15);
}

// K-loop + epilogue for one 16x16 tile.  GUARD=false: fully interior tile,
// no exec manipulation, no zero fills; GUARD=true: ragged edge tile.
template <bool GUARD>
__device__ inline void gemm_tile(const half_t* __restrict__ ap,
                                 const half_t* __restrict__ bp,
                                 float* __restrict__ Cf, half_t* __restrict__ Ch,
                                 const float* __restrict__ bias,
                                 int M, int N, int K, int ldc, long coff,
                                 int tileM, int tileN, int r, int g,
                                 bool aok, bool bok, float alpha, int act)
{
    v8f acc = {};
    int k0 = 0;
    for (; k0 + 64 <= K; k0 += 64) {
        if ((!GUARD || aok) && (k0 + 128) <= K) __builtin_prefetch(ap + k0 + 128, 0, 0);
        v16h a0 = {}, a1 = {}, b0 = {}, b1 = {};
        if (!GUARD || aok) { a0 = load_frag(ap + k0); a1 = load_frag(ap + k0 + 32); }
        if (!GUARD || bok) { b0 = load_frag(bp + k0); b1 = load_frag(bp + k0 + 32); }
        acc = __builtin_amdgcn_wmma_f32_16x16x32_f16(false, a0, false, b0,
                                                     (short)0, acc, false, false);
        acc = __builtin_amdgcn_wmma_f32_16x16x32_f16(false, a1, false, b1,
                                                     (short)0, acc, false, false);
    }
    for (; k0 < K; k0 += 32) {
        v16h a0 = {}, b0 = {};
        if (!GUARD || aok) a0 = load_frag(ap + k0);
        if (!GUARD || bok) b0 = load_frag(bp + k0);
        acc = __builtin_amdgcn_wmma_f32_16x16x32_f16(false, a0, false, b0,
                                                     (short)0, acc, false, false);
    }
    // C/D layout: lane (r,g), vgpr i -> (m = tileM + i + 8g, n = tileN + r)
    int n = tileN + r;
#pragma unroll
    for (int i = 0; i < 8; ++i) {
        int m = tileM + i + (g << 3);
        if (!GUARD || (m < M && n < N)) {
            float v = acc[i] * alpha;
            if (bias) v += bias[n];
            if (act)  v  = fmaxf(v, 0.f);
            if (Cf) Cf[coff + (long)m * ldc + n] = v;
            else    Ch[coff + (long)m * ldc + n] = (half_t)v;
        }
    }
}

// ---------------------------------------------------------------------------
// Batched WMMA GEMM:  out = act(alpha * A[M,K] * B^T + bias)
//   A: f16 [M,K] lda;  B: f16 [N,K] ldb (weight layout)  -- both 16B-aligned,
//   K % 32 == 0.  Output to Cf (f32) or Ch (f16), exactly one non-null.
// z-batch: z1=z/zdiv, z2=z%zdiv; base += z1*s?1 + z2*s?2 (element strides).
// 128 threads = 4 waves; each wave owns one 16x16 tile along N.
// ---------------------------------------------------------------------------
__global__ __launch_bounds__(128) void wmma_gemm_kernel(
    const half_t* __restrict__ A, const half_t* __restrict__ Bm,
    float* __restrict__ Cf, half_t* __restrict__ Ch,
    const float* __restrict__ bias,
    int M, int N, int K, int lda, int ldb, int ldc,
    float alpha, int act,
    long sA1, long sA2, long sB1, long sB2, long sC1, long sC2, int zdiv)
{
    int z  = blockIdx.z;
    int z1 = z / zdiv, z2 = z - z1 * zdiv;
    A  += z1 * sA1 + z2 * sA2;
    Bm += z1 * sB1 + z2 * sB2;
    long coff = z1 * sC1 + z2 * sC2;

    int wave  = threadIdx.x >> 5;
    int lane  = threadIdx.x & 31;
    int tileM = blockIdx.y << 4;
    int tileN = ((blockIdx.x << 2) + wave) << 4;
    if (tileN >= N) return;

    int r = lane & 15;
    int g = lane >> 4;
    int am = tileM + r;
    int bn = tileN + r;
    bool aok = am < M;
    bool bok = bn < N;
    const half_t* ap = A + (long)am * lda + (g << 3);
    const half_t* bp = Bm + (long)bn * ldb + (g << 3);

    // Uniform (wave-scalar) interior test -> single s_branch, no divergence.
    if ((tileM + 16 <= M) && (tileN + 16 <= N))
        gemm_tile<false>(ap, bp, Cf, Ch, bias, M, N, K, ldc, coff,
                         tileM, tileN, r, g, aok, bok, alpha, act);
    else
        gemm_tile<true>(ap, bp, Cf, Ch, bias, M, N, K, ldc, coff,
                        tileM, tileN, r, g, aok, bok, alpha, act);
}

// ---------------------------------------------------------------------------
// Conv front-end
// ---------------------------------------------------------------------------
__global__ __launch_bounds__(256) void conv1_kernel(const float* __restrict__ x,
                                                    const float* __restrict__ w,
                                                    const float* __restrict__ b,
                                                    float* __restrict__ y)
{
    int idx = blockIdx.x * 256 + threadIdx.x;           // [B, 384, 1500]
    if (idx >= BB * DD * TAUD) return;
    int t  = idx % TAUD;
    int o  = (idx / TAUD) % DD;
    int bb = idx / (TAUD * DD);
    const float* xb = x + (long)bb * 80 * TAUD;
    const float* wo = w + (long)o * 80 * 3;
    float acc = b[o];
    for (int i = 0; i < 80; ++i) {
        const float* xi = xb + (long)i * TAUD;
        const float* wi = wo + i * 3;
        if (t - 1 >= 0)   acc += xi[t - 1] * wi[0];
                          acc += xi[t]     * wi[1];
        if (t + 1 < TAUD) acc += xi[t + 1] * wi[2];
    }
    y[idx] = fmaxf(acc, 0.f);
}

__global__ __launch_bounds__(256) void conv2_kernel(const float* __restrict__ x,
                                                    const float* __restrict__ w,
                                                    const float* __restrict__ b,
                                                    float* __restrict__ y)
{
    int idx = blockIdx.x * 256 + threadIdx.x;           // out transposed: [B, 750, 384]
    if (idx >= BB * LENC * DD) return;
    int o  = idx % DD;
    int l  = (idx / DD) % LENC;
    int bb = idx / (DD * LENC);
    const float* xb = x + (long)bb * DD * TAUD;
    const float* wo = w + (long)o * DD * 3;
    int t0 = 2 * l - 1;
    float acc = b[o];
    for (int i = 0; i < DD; ++i) {
        const float* xi = xb + (long)i * TAUD;
        const float* wi = wo + i * 3;
        if (t0 >= 0) acc += xi[t0] * wi[0];
        acc += xi[t0 + 1] * wi[1];
        acc += xi[t0 + 2] * wi[2];
    }
    y[idx] = fmaxf(acc, 0.f);
}

// ---------------------------------------------------------------------------
// Row softmax, f32 in -> f16 out with K-padding to ldy (pad cols = 0).
// ---------------------------------------------------------------------------
__global__ __launch_bounds__(256) void softmax_f16pad_kernel(const float* __restrict__ x,
                                                             half_t* __restrict__ y,
                                                             int n, int ldy)
{
    long row = blockIdx.x;
    const float* xr = x + row * (long)n;
    half_t* yr = y + row * (long)ldy;
    __shared__ float red[256];
    int tid = threadIdx.x;
    float m = -1e30f;
    for (int i = tid; i < n; i += 256) m = fmaxf(m, xr[i]);
    red[tid] = m; __syncthreads();
    for (int o = 128; o > 0; o >>= 1) { if (tid < o) red[tid] = fmaxf(red[tid], red[tid + o]); __syncthreads(); }
    m = red[0]; __syncthreads();
    float s = 0.f;
    for (int i = tid; i < n; i += 256) s += __expf(xr[i] - m);
    red[tid] = s; __syncthreads();
    for (int o = 128; o > 0; o >>= 1) { if (tid < o) red[tid] += red[tid + o]; __syncthreads(); }
    float inv = 1.f / red[0];
    for (int i = tid; i < ldy; i += 256)
        yr[i] = (i < n) ? (half_t)(__expf(xr[i] - m) * inv) : (half_t)0.f;
}

// ---------------------------------------------------------------------------
// Build padded V^T: vt[((b*NHH+h)*HDIM+n)*LPAD + m] = V[b,m,h,n] (0 for m>=LENC)
// ---------------------------------------------------------------------------
__global__ __launch_bounds__(256) void build_vt_kernel(const half_t* __restrict__ qkv16,
                                                       half_t* __restrict__ vt)
{
    long idx = (long)blockIdx.x * 256 + threadIdx.x;
    long total = (long)BB * NHH * HDIM * LPAD;
    if (idx >= total) return;
    int  m    = idx % LPAD;
    long rest = idx / LPAD;
    int  n = rest % HDIM; rest /= HDIM;
    int  h = rest % NHH;
    int  b = rest / NHH;
    half_t v = (half_t)0.f;
    if (m < LENC)
        v = qkv16[((long)(b * LENC + m)) * (3 * DD) + 2 * DD + h * HDIM + n];
    vt[idx] = v;
}

// ---------------------------------------------------------------------------
// x = LayerNorm(x + y) * g + b ; also emits f16 copy. One block per row.
// ---------------------------------------------------------------------------
__global__ __launch_bounds__(256) void ln_res_kernel(float* __restrict__ x,
                                                     const float* __restrict__ y,
                                                     const float* __restrict__ g,
                                                     const float* __restrict__ b,
                                                     half_t* __restrict__ x16, int D_)
{
    long row = blockIdx.x;
    int  tid = threadIdx.x;
    __shared__ float red[256];
    float* xr = x + row * (long)D_;
    const float* yr = y + row * (long)D_;
    float s0 = 0.f, s1 = 0.f;
    for (int d = tid; d < D_; d += 256) { float s = xr[d] + yr[d]; s0 += s; s1 += s * s; }
    red[tid] = s0; __syncthreads();
    for (int o = 128; o > 0; o >>= 1) { if (tid < o) red[tid] += red[tid + o]; __syncthreads(); }
    float mean = red[0] / D_; __syncthreads();
    red[tid] = s1; __syncthreads();
    for (int o = 128; o > 0; o >>= 1) { if (tid < o) red[tid] += red[tid + o]; __syncthreads(); }
    float var = red[0] / D_ - mean * mean;
    float inv = rsqrtf(var + 1e-5f);
    for (int d = tid; d < D_; d += 256) {
        float s = xr[d] + yr[d];
        float v = (s - mean) * inv * g[d] + b[d];
        xr[d] = v;
        if (x16) x16[row * (long)D_ + d] = (half_t)v;
    }
}

// ---------------------------------------------------------------------------
// Decoder matvecs:  out[b,j] = bias[j] + sum_k x[b,k] * W[j,k]
// ---------------------------------------------------------------------------
__global__ __launch_bounds__(256) void matvec_kernel(float* __restrict__ out,
                                                     const float* __restrict__ x,
                                                     const float* __restrict__ W,
                                                     const float* __restrict__ bias,
                                                     int Bn, int N, int K)
{
    long idx = (long)blockIdx.x * 256 + threadIdx.x;
    if (idx >= (long)Bn * N) return;
    int b = idx / N, j = idx % N;
    const float* xr = x + (long)b * K;
    const float* wr = W + (long)j * K;
    float acc = bias ? bias[j] : 0.f;
    for (int k = 0; k < K; ++k) acc += xr[k] * wr[k];
    out[idx] = acc;
}

// x = concat(x1[b, 0..K1), emb[token[b], 0..384))
__global__ __launch_bounds__(256) void matvec_cat_kernel(float* __restrict__ out,
                                                         const float* __restrict__ x1, int K1,
                                                         const float* __restrict__ emb,
                                                         const int* __restrict__ token,
                                                         const float* __restrict__ W,
                                                         const float* __restrict__ bias,
                                                         int Bn, int N, int Ktot)
{
    long idx = (long)blockIdx.x * 256 + threadIdx.x;
    if (idx >= (long)Bn * N) return;
    int b = idx / N, j = idx % N;
    const float* wr = W + (long)j * Ktot;
    const float* xr = x1 + (long)b * K1;
    const float* er = emb + (long)token[b] * DD;
    float acc = bias[j];
    for (int k = 0; k < K1; ++k)    acc += xr[k] * wr[k];
    for (int k = K1; k < Ktot; ++k) acc += er[k - K1] * wr[k];
    out[idx] = acc;
}

__global__ __launch_bounds__(256) void gru_combine_kernel(float* __restrict__ hnew,
                                                          const float* __restrict__ hold,
                                                          const float* __restrict__ gi,
                                                          const float* __restrict__ gh,
                                                          int Bn, int H)
{
    int idx = blockIdx.x * 256 + threadIdx.x;
    if (idx >= Bn * H) return;
    int b = idx / H, i = idx % H;
    const float* gib = gi + (long)b * 3 * H;
    const float* ghb = gh + (long)b * 3 * H;
    float r = 1.f / (1.f + __expf(-(gib[i] + ghb[i])));
    float z = 1.f / (1.f + __expf(-(gib[H + i] + ghb[H + i])));
    float n = tanhf(gib[2 * H + i] + r * ghb[2 * H + i]);
    hnew[idx] = (1.f - z) * n + z * hold[idx];
}

// v_la[b,l] = enc_asr[b,l,:].la_v_w + la_in_b[2]
__global__ __launch_bounds__(256) void vla_kernel(float* __restrict__ vla,
                                                  const float* __restrict__ encA,
                                                  const float* __restrict__ la_v_w,
                                                  const float* __restrict__ la_in_b)
{
    int idx = blockIdx.x * 256 + threadIdx.x;
    if (idx >= BB * LENC) return;
    const float* e = encA + (long)idx * DD;
    float acc = la_in_b[2];
    for (int d = 0; d < DD; ++d) acc += e[d] * la_v_w[d];
    vla[idx] = acc;
}

// ---------------------------------------------------------------------------
// Location attention step: s[l,m] = q_l * k_m is rank-1, row max is
// q_l * (q_l>=0 ? max(k) : min(k)); k/v cached in LDS, one block per batch.
// ---------------------------------------------------------------------------
__global__ __launch_bounds__(256) void loc_attn_kernel(const float* __restrict__ h_asr,
                                                       const float* __restrict__ attns_old,
                                                       const float* __restrict__ vla,
                                                       float* __restrict__ attns_new,
                                                       const float* __restrict__ la_q_w,
                                                       const float* __restrict__ la_k_w,
                                                       const float* __restrict__ la_in_b,
                                                       const float* __restrict__ la_out_w,
                                                       const float* __restrict__ la_out_b)
{
    int b = blockIdx.x, tid = threadIdx.x;
    __shared__ float kk[LENC], vv[LENC];
    __shared__ float red[256], red2[256];
    float laq = la_q_w[0], lak = la_k_w[0];
    float b0 = la_in_b[0], b1 = la_in_b[1];
    float low = la_out_w[0], lob = la_out_b[0];
    float mx = -1e30f, mn = 1e30f;
    for (int m = tid; m < LENC; m += 256) {
        float kv = attns_old[b * LENC + m] * lak + b1;
        kk[m] = kv; vv[m] = vla[b * LENC + m];
        mx = fmaxf(mx, kv); mn = fminf(mn, kv);
    }
    red[tid] = mx; red2[tid] = mn; __syncthreads();
    for (int o = 128; o > 0; o >>= 1) {
        if (tid < o) { red[tid] = fmaxf(red[tid], red[tid + o]); red2[tid] = fminf(red2[tid], red2[tid + o]); }
        __syncthreads();
    }
    float kmax = red[0], kmin = red2[0];
    __syncthreads();
    for (int l = tid; l < LENC; l += 256) {
        float q    = h_asr[b * LENC + l] * laq + b0;
        float smax = (q >= 0.f) ? q * kmax : q * kmin;
        float sum = 0.f, ws = 0.f;
        for (int m = 0; m < LENC; ++m) {
            float e = __expf(q * kk[m] - smax);
            sum += e; ws += e * vv[m];
        }
        attns_new[b * LENC + l] = low * (ws / sum) + lob;
    }
}

// asr_ctx[b,d] = sum_l encA[b,l,d] w[b,l] ; spk_ctx same with encD
__global__ __launch_bounds__(256) void ctx_einsum_kernel(float* __restrict__ actx,
                                                         float* __restrict__ sctx,
                                                         const float* __restrict__ encA,
                                                         const float* __restrict__ encD,
                                                         const float* __restrict__ w)
{
    int idx = blockIdx.x * 256 + threadIdx.x;
    if (idx >= BB * DD) return;
    int b = idx / DD, d = idx % DD;
    const float* wb = w + b * LENC;
    const float* ea = encA + (long)b * LENC * DD + d;
    const float* ed = encD + (long)b * LENC * DD + d;
    float a = 0.f, s = 0.f;
    for (int l = 0; l < LENC; ++l) { float ww = wb[l]; a += ea[(long)l * DD] * ww; s += ed[(long)l * DD] * ww; }
    actx[idx] = a; sctx[idx] = s;
}

// ---------------------------------------------------------------------------
// Speaker attention: beta = softmax(<inv,h>/(|h||inv|)); spk_state = inv^T beta
// ---------------------------------------------------------------------------
__global__ __launch_bounds__(256) void spk_attn_kernel(const float* __restrict__ inv,
                                                       const float* __restrict__ h,
                                                       float* __restrict__ spk_state,
                                                       float* __restrict__ dout_diar, int t)
{
    int b = blockIdx.x, tid = threadIdx.x;
    __shared__ float red[256], rdot[256], rn[256];
    __shared__ float sc[SSPK], nn[SSPK], beta[SSPK];
    __shared__ float hn;
    float s = 0.f;
    for (int d = tid; d < DD; d += 256) { float v = h[b * DD + d]; s += v * v; }
    red[tid] = s; __syncthreads();
    for (int o = 128; o > 0; o >>= 1) { if (tid < o) red[tid] += red[tid + o]; __syncthreads(); }
    if (tid == 0) hn = sqrtf(red[0]);
    int sIdx = tid >> 5, ln = tid & 31;
    float dot = 0.f, nsum = 0.f;
    for (int d = ln; d < DD; d += 32) {
        float w = inv[((long)b * SSPK + sIdx) * DD + d];
        dot += w * h[b * DD + d]; nsum += w * w;
    }
    rdot[tid] = dot; rn[tid] = nsum; __syncthreads();
    for (int o = 16; o > 0; o >>= 1) {
        if (ln < o) { rdot[tid] += rdot[tid + o]; rn[tid] += rn[tid + o]; }
        __syncthreads();
    }
    if (ln == 0) { sc[sIdx] = rdot[tid]; nn[sIdx] = sqrtf(rn[tid]); }
    __syncthreads();
    if (tid == 0) {
        float e[SSPK], mxv = -1e30f;
        for (int q = 0; q < SSPK; ++q) { float v = sc[q] / (hn * nn[q]); e[q] = v; mxv = fmaxf(mxv, v); }
        float sum = 0.f;
        for (int q = 0; q < SSPK; ++q) { e[q] = __expf(e[q] - mxv); sum += e[q]; }
        for (int q = 0; q < SSPK; ++q) {
            beta[q] = e[q] / sum;
            dout_diar[((long)b * TT + t) * SSPK + q] = beta[q];
        }
    }
    __syncthreads();
    for (int d = tid; d < DD; d += 256) {
        float acc = 0.f;
        for (int q = 0; q < SSPK; ++q) acc += inv[((long)b * SSPK + q) * DD + d] * beta[q];
        spk_state[b * DD + d] = acc;
    }
}

// pre[b,d] = asr_ctx + h_asr@h2a^T + h2a_b + spk@d2a^T + d2a_b
__global__ __launch_bounds__(256) void pre_kernel(float* __restrict__ pre,
                                                  const float* __restrict__ actx,
                                                  const float* __restrict__ h_asr,
                                                  const float* __restrict__ spk,
                                                  const float* __restrict__ h2a_w,
                                                  const float* __restrict__ h2a_b,
                                                  const float* __restrict__ d2a_w,
                                                  const float* __restrict__ d2a_b)
{
    int idx = blockIdx.x * 256 + threadIdx.x;
    if (idx >= BB * DD) return;
    int b = idx / DD, d = idx % DD;
    float acc = actx[idx] + h2a_b[d] + d2a_b[d];
    const float* hw = h2a_w + (long)d * LENC;
    const float* hb = h_asr + b * LENC;
    for (int l = 0; l < LENC; ++l) acc += hb[l] * hw[l];
    const float* dw = d2a_w + (long)d * DD;
    const float* sb = spk + b * DD;
    for (int k = 0; k < DD; ++k) acc += sb[k] * dw[k];
    pre[idx] = acc;
}

__global__ __launch_bounds__(256) void logits_kernel(float* __restrict__ out_asr,
                                                     const float* __restrict__ pre,
                                                     const float* __restrict__ w,
                                                     const float* __restrict__ bias, int t)
{
    int idx = blockIdx.x * 256 + threadIdx.x;
    if (idx >= BB * VOC) return;
    int b = idx / VOC, v = idx % VOC;
    const float* pb = pre + b * DD;
    const float* wr = w + (long)v * DD;
    float acc = bias[v];
    for (int d = 0; d < DD; ++d) acc += pb[d] * wr[d];
    out_asr[((long)b * TT + t) * VOC + v] = acc;
}

__global__ void token_kernel(int* __restrict__ token, const int* __restrict__ tgt, int t)
{
    int b = threadIdx.x;
    if (b < BB) { int v = tgt[b * TT + t]; token[b] = (v == -100) ? 0 : v; }
}

// loss = nll(asr) + nll(diar) on raw scores, ignore_index=-100
__global__ __launch_bounds__(256) void loss_kernel(const float* __restrict__ outA,
                                                   const float* __restrict__ outD,
                                                   const int* __restrict__ tA,
                                                   const int* __restrict__ tD,
                                                   float* __restrict__ loss)
{
    __shared__ float r0[256], r1[256];
    __shared__ int   c0[256], c1[256];
    int tid = threadIdx.x;
    float sA = 0.f, sD = 0.f; int cA = 0, cD = 0;
    for (int i = tid; i < BB * TT; i += 256) {
        int va = tA[i]; if (va != -100) { sA += outA[(long)i * VOC + va]; cA++; }
        int vd = tD[i]; if (vd != -100) { sD += outD[(long)i * SSPK + vd]; cD++; }
    }
    r0[tid] = sA; r1[tid] = sD; c0[tid] = cA; c1[tid] = cD; __syncthreads();
    for (int o = 128; o > 0; o >>= 1) {
        if (tid < o) { r0[tid] += r0[tid + o]; r1[tid] += r1[tid + o]; c0[tid] += c0[tid + o]; c1[tid] += c1[tid + o]; }
        __syncthreads();
    }
    if (tid == 0)
        loss[0] = -(r0[0] / fmaxf((float)c0[0], 1.f)) - (r1[0] / fmaxf((float)c1[0], 1.f));
}

// ---------------------------------------------------------------------------
// Host orchestration
// ---------------------------------------------------------------------------
static void launch_gemm(hipStream_t st, const half_t* A, const half_t* B,
                        float* Cf, half_t* Ch, const float* bias,
                        int M, int N, int K, int lda, int ldb, int ldc,
                        float alpha, int act,
                        long sA1, long sA2, long sB1, long sB2, long sC1, long sC2,
                        int zdiv, int zcnt)
{
    dim3 grid((N + 63) / 64, (M + 15) / 16, zcnt);
    wmma_gemm_kernel<<<grid, dim3(128), 0, st>>>(A, B, Cf, Ch, bias, M, N, K, lda, ldb, ldc,
                                                 alpha, act, sA1, sA2, sB1, sB2, sC1, sC2, zdiv);
}

extern "C" void kernel_launch(void* const* d_in, const int* in_sizes, int n_in,
                              void* d_out, int out_size, void* d_ws, size_t ws_size,
                              hipStream_t stream)
{
    (void)in_sizes; (void)n_in; (void)out_size; (void)ws_size;
    // ---- inputs -----------------------------------------------------------
    const float* in_feat  = (const float*)d_in[0];
    const float* spk_inv  = (const float*)d_in[1];
    const float* conv1_w  = (const float*)d_in[2];
    const float* conv1_b  = (const float*)d_in[3];
    const float* conv2_w  = (const float*)d_in[4];
    const float* conv2_b  = (const float*)d_in[5];
    const float* encW[2][12];   // [stack][qkv_w,qkv_b,ow,ob,f1w,f1b,f2w,f2b,g1,b1,g2,b2]
    for (int s = 0; s < 2; ++s)
        for (int k = 0; k < 12; ++k) encW[s][k] = (const float*)d_in[6 + s * 12 + k];
    const float* la_q_w   = (const float*)d_in[30];
    const float* la_k_w   = (const float*)d_in[31];
    const float* la_v_w   = (const float*)d_in[32];
    const float* la_in_b  = (const float*)d_in[33];
    const float* la_out_w = (const float*)d_in[34];
    const float* la_out_b = (const float*)d_in[35];
    const float* emb      = (const float*)d_in[36];
    const float* a_wih    = (const float*)d_in[37];
    const float* a_whh    = (const float*)d_in[38];
    const float* a_bih    = (const float*)d_in[39];
    const float* a_bhh    = (const float*)d_in[40];
    const float* d_wih    = (const float*)d_in[41];
    const float* d_whh    = (const float*)d_in[42];
    const float* d_bih    = (const float*)d_in[43];
    const float* d_bhh    = (const float*)d_in[44];
    const float* asr_out_w = (const float*)d_in[45];
    const float* asr_out_b = (const float*)d_in[46];
    const float* d2a_w    = (const float*)d_in[47];
    const float* d2a_b    = (const float*)d_in[48];
    const float* h2a_w    = (const float*)d_in[49];
    const float* h2a_b    = (const float*)d_in[50];
    const int*   tgtA     = (const int*)d_in[51];
    const int*   tgtD     = (const int*)d_in[52];

    float* out      = (float*)d_out;
    float* out_asr  = out;                      // [8,256,60]
    float* out_diar = out + BB * TT * VOC;      // [8,256,8]
    float* out_loss = out + BB * TT * VOC + BB * TT * SSPK;

    // ---- workspace bump allocator ----------------------------------------
    char*  base = (char*)d_ws;
    size_t off  = 0;
    auto alloc = [&](size_t bytes) -> void* {
        void* p = base + off;
        off += (bytes + 255) & ~(size_t)255;
        return p;
    };
    float*  xcv1   = (float*)alloc((size_t)BB * DD * TAUD * 4);
    float*  x0     = (float*)alloc((size_t)MROW * DD * 4);
    float*  encA   = (float*)alloc((size_t)MROW * DD * 4);
    float*  encD   = (float*)alloc((size_t)MROW * DD * 4);
    half_t* h16a   = (half_t*)alloc((size_t)MROW * DD * 2);              // x f16
    half_t* h16b   = (half_t*)alloc((size_t)MROW * 3 * DD * 2);          // qkv f16
    float*  scores = (float*)alloc((size_t)BB * NHH * LENC * LENC * 4);
    half_t* sco16  = (half_t*)alloc((size_t)BB * NHH * LENC * LPAD * 2); // softmax, K-padded
    half_t* vt     = (half_t*)alloc((size_t)BB * NHH * HDIM * LPAD * 2); // V^T, K-padded
    half_t* ctx16  = (half_t*)alloc((size_t)MROW * DD * 2);
    half_t* ffh16  = (half_t*)alloc((size_t)MROW * FFD * 2);
    float*  tmp    = (float*)alloc((size_t)MROW * DD * 4);
    half_t* wq16[2], *wo16[2], *w116[2], *w216[2];
    for (int s = 0; s < 2; ++s) {
        wq16[s] = (half_t*)alloc((size_t)NLAY * 3 * DD * DD * 2);
        wo16[s] = (half_t*)alloc((size_t)NLAY * DD * DD * 2);
        w116[s] = (half_t*)alloc((size_t)NLAY * FFD * DD * 2);
        w216[s] = (half_t*)alloc((size_t)NLAY * DD * FFD * 2);
    }
    float* vla     = (float*)alloc((size_t)BB * LENC * 4);
    float* actx[2] = { (float*)alloc(BB * DD * 4), (float*)alloc(BB * DD * 4) };
    float* hasr[2] = { (float*)alloc(BB * LENC * 4), (float*)alloc(BB * LENC * 4) };
    float* hdia[2] = { (float*)alloc(BB * DD * 4), (float*)alloc(BB * DD * 4) };
    float* attn[2] = { (float*)alloc(BB * LENC * 4), (float*)alloc(BB * LENC * 4) };
    int*   token   = (int*)alloc(BB * 4);
    float* gi_a    = (float*)alloc((size_t)BB * 3 * LENC * 4);
    float* gh_a    = (float*)alloc((size_t)BB * 3 * LENC * 4);
    float* gi_d    = (float*)alloc((size_t)BB * 3 * DD * 4);
    float* gh_d    = (float*)alloc((size_t)BB * 3 * DD * 4);
    float* spk_ctx = (float*)alloc(BB * DD * 4);
    float* spk_st  = (float*)alloc(BB * DD * 4);
    float* pre     = (float*)alloc(BB * DD * 4);

    auto cvt = [&](const float* src, half_t* dst, long n) {
        long blocks = (n + 255) / 256;
        cvt_f16_kernel<<<dim3((unsigned)blocks), dim3(256), 0, stream>>>(src, dst, n);
    };

    // ---- convert encoder weights to f16 ----------------------------------
    for (int s = 0; s < 2; ++s) {
        cvt(encW[s][0], wq16[s], (long)NLAY * 3 * DD * DD);
        cvt(encW[s][2], wo16[s], (long)NLAY * DD * DD);
        cvt(encW[s][4], w116[s], (long)NLAY * FFD * DD);
        cvt(encW[s][6], w216[s], (long)NLAY * DD * FFD);
    }

    // ---- conv front-end ---------------------------------------------------
    conv1_kernel<<<dim3((BB * DD * TAUD + 255) / 256), dim3(256), 0, stream>>>(in_feat, conv1_w, conv1_b, xcv1);
    conv2_kernel<<<dim3((BB * LENC * DD + 255) / 256), dim3(256), 0, stream>>>(xcv1, conv2_w, conv2_b, x0);
    hipMemcpyAsync(encA, x0, (size_t)MROW * DD * 4, hipMemcpyDeviceToDevice, stream);
    hipMemcpyAsync(encD, x0, (size_t)MROW * DD * 4, hipMemcpyDeviceToDevice, stream);

    // ---- encoder stacks (WMMA GEMMs, all K % 32 == 0, 16B-aligned rows) ---
    const float inv_sqrt_hd = 0.1020620726f; // 1/sqrt(96)
    auto enc_layer = [&](float* X, int s, int l) {
        const float* qkv_b = encW[s][1] + l * 3 * DD;
        const float* ob    = encW[s][3] + l * DD;
        const float* f1b   = encW[s][5] + l * FFD;
        const float* f2b   = encW[s][7] + l * DD;
        const float* g1    = encW[s][8] + l * DD;
        const float* b1    = encW[s][9] + l * DD;
        const float* g2    = encW[s][10] + l * DD;
        const float* b2    = encW[s][11] + l * DD;
        const half_t* wq = wq16[s] + (long)l * 3 * DD * DD;
        const half_t* wo = wo16[s] + (long)l * DD * DD;
        const half_t* w1 = w116[s] + (long)l * FFD * DD;
        const half_t* w2 = w216[s] + (long)l * DD * FFD;

        // qkv (f16 out)
        launch_gemm(stream, h16a, wq, nullptr, h16b, qkv_b, MROW, 3 * DD, DD, DD, DD, 3 * DD,
                    1.f, 0, 0, 0, 0, 0, 0, 0, 1, 1);
        // scores[b,h,l,m] = Q.K^T / sqrt(hd)  (f32 out)
        launch_gemm(stream, h16b, h16b + DD, scores, nullptr, nullptr, LENC, LENC, HDIM,
                    3 * DD, 3 * DD, LENC, inv_sqrt_hd, 0,
                    (long)LENC * 3 * DD, HDIM, (long)LENC * 3 * DD, HDIM,
                    (long)NHH * LENC * LENC, (long)LENC * LENC, NHH, BB * NHH);
        // softmax -> f16, K-padded to 768
        softmax_f16pad_kernel<<<dim3(BB * NHH * LENC), dim3(256), 0, stream>>>(scores, sco16, LENC, LPAD);
        // V^T, K-padded
        build_vt_kernel<<<dim3((unsigned)(((long)BB * NHH * HDIM * LPAD + 255) / 256)), dim3(256), 0, stream>>>(h16b, vt);
        // ctx[b,l,h*96+n] = A.V  (f16 out)
        launch_gemm(stream, sco16, vt, nullptr, ctx16, nullptr, LENC, HDIM, LPAD,
                    LPAD, LPAD, DD, 1.f, 0,
                    (long)NHH * LENC * LPAD, (long)LENC * LPAD,
                    (long)NHH * HDIM * LPAD, (long)HDIM * LPAD,
                    (long)LENC * DD, HDIM, NHH, BB * NHH);
        // out-proj (f32 out)
        launch_gemm(stream, ctx16, wo, tmp, nullptr, ob, MROW, DD, DD, DD, DD, DD,
                    1.f, 0, 0, 0, 0, 0, 0, 0, 1, 1);
        ln_res_kernel<<<dim3(MROW), dim3(256), 0, stream>>>(X, tmp, g1, b1, h16a, DD);
        // FF1 (relu, f16 out)
        launch_gemm(stream, h16a, w1, nullptr, ffh16, f1b, MROW, FFD, DD, DD, DD, FFD,
                    1.f, 1, 0, 0, 0, 0, 0, 0, 1, 1);
        // FF2 (f32 out)
        launch_gemm(stream, ffh16, w2, tmp, nullptr, f2b, MROW, DD, FFD, FFD, FFD, DD,
                    1.f, 0, 0, 0, 0, 0, 0, 0, 1, 1);
        ln_res_kernel<<<dim3(MROW), dim3(256), 0, stream>>>(X, tmp, g2, b2, h16a, DD);
    };
    for (int s = 0; s < 2; ++s) {
        float* X = s ? encD : encA;
        cvt(X, h16a, (long)MROW * DD);
        for (int l = 0; l < NLAY; ++l) enc_layer(X, s, l);
    }

    // ---- decoder ----------------------------------------------------------
    vla_kernel<<<dim3((BB * LENC + 255) / 256), dim3(256), 0, stream>>>(vla, encA, la_v_w, la_in_b);
    hipMemsetAsync(actx[0], 0, BB * DD * 4, stream);
    hipMemsetAsync(hasr[0], 0, BB * LENC * 4, stream);
    hipMemsetAsync(hdia[0], 0, BB * DD * 4, stream);
    hipMemsetAsync(attn[0], 0, BB * LENC * 4, stream);
    hipMemsetAsync(token, 0, BB * 4, stream);

    for (int t = 0; t < TT; ++t) {
        int c = t & 1, w = 1 - c;
        // ASR GRU (hidden = 750)
        matvec_cat_kernel<<<dim3((BB * 3 * LENC + 255) / 256), dim3(256), 0, stream>>>(
            gi_a, actx[c], DD, emb, token, a_wih, a_bih, BB, 3 * LENC, 2 * DD);
        matvec_kernel<<<dim3((BB * 3 * LENC + 255) / 256), dim3(256), 0, stream>>>(
            gh_a, hasr[c], a_whh, a_bhh, BB, 3 * LENC, LENC);
        gru_combine_kernel<<<dim3((BB * LENC + 255) / 256), dim3(256), 0, stream>>>(
            hasr[w], hasr[c], gi_a, gh_a, BB, LENC);
        // location attention
        loc_attn_kernel<<<dim3(BB), dim3(256), 0, stream>>>(
            hasr[w], attn[c], vla, attn[w], la_q_w, la_k_w, la_in_b, la_out_w, la_out_b);
        // contexts
        ctx_einsum_kernel<<<dim3((BB * DD + 255) / 256), dim3(256), 0, stream>>>(
            actx[w], spk_ctx, encA, encD, attn[w]);
        // diar GRU (hidden = 384)
        matvec_cat_kernel<<<dim3((BB * 3 * DD + 255) / 256), dim3(256), 0, stream>>>(
            gi_d, spk_ctx, DD, emb, token, d_wih, d_bih, BB, 3 * DD, 2 * DD);
        matvec_kernel<<<dim3((BB * 3 * DD + 255) / 256), dim3(256), 0, stream>>>(
            gh_d, hdia[c], d_whh, d_bhh, BB, 3 * DD, DD);
        gru_combine_kernel<<<dim3((BB * DD + 255) / 256), dim3(256), 0, stream>>>(
            hdia[w], hdia[c], gi_d, gh_d, BB, DD);
        // speaker attention -> beta (diar output) + spk_state
        spk_attn_kernel<<<dim3(BB), dim3(256), 0, stream>>>(spk_inv, hdia[w], spk_st, out_diar, t);
        // logits
        pre_kernel<<<dim3((BB * DD + 255) / 256), dim3(256), 0, stream>>>(
            pre, actx[w], hasr[w], spk_st, h2a_w, h2a_b, d2a_w, d2a_b);
        logits_kernel<<<dim3((BB * VOC + 255) / 256), dim3(256), 0, stream>>>(
            out_asr, pre, asr_out_w, asr_out_b, t);
        // next token (teacher forcing)
        token_kernel<<<dim3(1), dim3(32), 0, stream>>>(token, tgtA, t);
    }

    // ---- loss -------------------------------------------------------------
    loss_kernel<<<dim3(1), dim3(256), 0, stream>>>(out_asr, out_diar, tgtA, tgtD, out_loss);
}
